// GCNNet_41120016892602
// MI455X (gfx1250) — matrix operand order repaired and checked by reading, not compiled, not run on hardware
//
#include <hip/hip_runtime.h>
#include <hip/hip_bf16.h>
#include <math.h>

typedef __attribute__((ext_vector_type(8)))  float  v8f;
typedef __attribute__((ext_vector_type(16))) __bf16 v16bf;
typedef __attribute__((ext_vector_type(8)))  __bf16 v8bf;
typedef __attribute__((ext_vector_type(2)))  __bf16 v2bf;

#define N_NODES  100000   /* == 6250 * 16 : every WMMA M-tile is full */
#define N_EDGES  3200000
#define N_GRAPHS 2000
#define IN_F     37
#define K1PAD    64       /* layer-1 K padded to 2 WMMA K-steps */
#define DIM      32
#define GDIM     128
#define CAT      (3 * DIM) /* 96 */

// ---------------------------------------------------------------- degree / norm
__global__ void k_deg(const int* __restrict__ dst, float* __restrict__ deg) {
  int i = blockIdx.x * blockDim.x + threadIdx.x;
  if (i < N_EDGES) atomicAdd(&deg[dst[i]], 1.0f);
}

__global__ void k_dinv(const float* __restrict__ deg, float* __restrict__ dinv) {
  int i = blockIdx.x * blockDim.x + threadIdx.x;
  if (i < N_NODES) dinv[i] = rsqrtf(deg[i] + 2.0f);
}

// -------------------------------------- pack x (fp32 [N,37]) -> bf16 [N,64] zero-padded
__global__ void k_pack_x(const float* __restrict__ x, __bf16* __restrict__ xp) {
  int idx = blockIdx.x * blockDim.x + threadIdx.x;   // N_NODES * 32, 2 bf16 per thread
  if (idx >= N_NODES * (K1PAD / 2)) return;
  int r = idx >> 5;
  int k = (idx & 31) * 2;
  v2bf p;
  p[0] = (__bf16)((k     < IN_F) ? x[(size_t)r * IN_F + k]     : 0.0f);
  p[1] = (__bf16)((k + 1 < IN_F) ? x[(size_t)r * IN_F + k + 1] : 0.0f);
  *(v2bf*)(xp + (size_t)r * K1PAD + k) = p;
}

// ------------------------------------------------- pack weights into WMMA B-frag order
// B fragment layout (v_wmma_f32_16x16x32_bf16, wave32):
//   lane<16 holds K=0..7,16..23, lane>=16 holds K=8..15,24..31 (e<8 -> 8h+e, e>=8 -> 16+8h+e-8), N = lane&15.
// Linear index: ((nt*Ksteps + ks)*32 + lane)*16 + e
__global__ void k_pack_b(const float* __restrict__ B, __bf16* __restrict__ out,
                         int Kreal, int Ncols, int Ksteps, int total) {
  int idx = blockIdx.x * blockDim.x + threadIdx.x;
  if (idx >= total) return;
  int e    = idx & 15;
  int t    = idx >> 4;
  int lane = t & 31; t >>= 5;
  int ks   = t % Ksteps;
  int nt   = t / Ksteps;
  int K = ks * 32 + ((lane >= 16) ? 8 : 0) + (e & 7) + ((e >= 8) ? 16 : 0);
  int n = nt * 16 + (lane & 15);
  float v = (K < Kreal) ? B[K * Ncols + n] : 0.0f;
  out[idx] = (__bf16)v;
}

// ------------------------------------------------------------------- WMMA GEMM (fp32 A)
// One wave -> full 16-row M-tile x (NT*16) cols.  Branch-free: LDA%4==0, K%32==0.
// A-frag per lane: two contiguous 8-float runs {kbase..+7},{kbase+16..+23}, cvt to bf16.
template <int NT, int LDA, int KSTEPS, bool BIAS, bool RELU>
__global__ void k_gemm_wmma(const float* __restrict__ A,
                            const __bf16* __restrict__ Bp, const float* __restrict__ bias,
                            float* __restrict__ C, int ldc, int Mrows) {
  static_assert((LDA & 3) == 0, "float4 A path requires LDA%4==0");
  const int lane = threadIdx.x & 31;
  const int wave = threadIdx.x >> 5;
  const int tile = blockIdx.x * (blockDim.x >> 5) + wave;
  const int m0 = tile << 4;
  if (m0 >= Mrows) return;                 // wave-uniform -> EXEC stays all-ones
  const int r = m0 + (lane & 15);
  const float* Arow = A + (size_t)r * LDA;
  const int hi = (lane >= 16) ? 8 : 0;

  v8f acc[NT] = {};

#pragma unroll
  for (int ks = 0; ks < KSTEPS; ++ks) {
    const int kbase = ks * 32 + hi;
    float tmp[16];
    const float4* p0 = (const float4*)(Arow + kbase);
    const float4* p1 = (const float4*)(Arow + kbase + 16);
    *(float4*)&tmp[0]  = p0[0];
    *(float4*)&tmp[4]  = p0[1];
    *(float4*)&tmp[8]  = p1[0];
    *(float4*)&tmp[12] = p1[1];
    v16bf a;
#pragma unroll
    for (int e = 0; e < 16; ++e) a[e] = (__bf16)tmp[e];
#pragma unroll
    for (int nt = 0; nt < NT; ++nt) {
      const v16bf b = *(const v16bf*)(Bp + ((((size_t)nt * KSTEPS + ks) * 32 + lane) << 4));
      acc[nt] = __builtin_amdgcn_wmma_f32_16x16x32_bf16(
          false, a, false, b, (short)0, acc[nt], false, false);
    }
  }

  const int ncol = lane & 15;
#pragma unroll
  for (int nt = 0; nt < NT; ++nt) {
    const int n = nt * 16 + ncol;
    const float badd = BIAS ? bias[n] : 0.0f;
#pragma unroll
    for (int j = 0; j < 8; ++j) {
      const int mr = m0 + hi + j;          // lanes<16: M=j ; lanes>=16: M=j+8 (full tile)
      float v = acc[nt][j] + badd;
      if (RELU) v = fmaxf(v, 0.0f);
      C[(size_t)mr * ldc + n] = v;
    }
  }
}

// --------------------------------------------------------- WMMA GEMM (pre-packed bf16 A)
// A is bf16 row-major [Mrows x KSTEPS*32] (zero-padded).  Used for layer 1 (K=37->64).
template <int NT, int KSTEPS>
__global__ void k_gemm_wmma_bf16A(const __bf16* __restrict__ A,
                                  const __bf16* __restrict__ Bp,
                                  float* __restrict__ C, int ldc, int Mrows) {
  const int lane = threadIdx.x & 31;
  const int wave = threadIdx.x >> 5;
  const int tile = blockIdx.x * (blockDim.x >> 5) + wave;
  const int m0 = tile << 4;
  if (m0 >= Mrows) return;
  const int r = m0 + (lane & 15);
  const __bf16* Arow = A + (size_t)r * (KSTEPS * 32);
  const int hi = (lane >= 16) ? 8 : 0;

  v8f acc[NT] = {};

#pragma unroll
  for (int ks = 0; ks < KSTEPS; ++ks) {
    const int kbase = ks * 32 + hi;
    const v8bf a0 = *(const v8bf*)(Arow + kbase);        // 16B load, K run {kbase..+7}
    const v8bf a1 = *(const v8bf*)(Arow + kbase + 16);   // 16B load, K run {kbase+16..+23}
    v16bf a;
#pragma unroll
    for (int e = 0; e < 8; ++e) { a[e] = a0[e]; a[e + 8] = a1[e]; }
#pragma unroll
    for (int nt = 0; nt < NT; ++nt) {
      const v16bf b = *(const v16bf*)(Bp + ((((size_t)nt * KSTEPS + ks) * 32 + lane) << 4));
      acc[nt] = __builtin_amdgcn_wmma_f32_16x16x32_bf16(
          false, a, false, b, (short)0, acc[nt], false, false);
    }
  }

  const int ncol = lane & 15;
#pragma unroll
  for (int nt = 0; nt < NT; ++nt) {
    const int n = nt * 16 + ncol;
#pragma unroll
    for (int j = 0; j < 8; ++j) {
      const int mr = m0 + hi + j;
      C[(size_t)mr * ldc + n] = acc[nt][j];
    }
  }
}

// ----------------------------------------------- GCN layer: self-loop init, edges, relu
__global__ void k_self_init(const float* __restrict__ h, const float* __restrict__ dinv,
                            const float* __restrict__ b, float* __restrict__ hcat,
                            int layerOff) {
  int idx = blockIdx.x * blockDim.x + threadIdx.x;
  if (idx >= N_NODES * DIM) return;
  int r = idx >> 5, c = idx & 31;
  float di = dinv[r];
  hcat[(size_t)r * CAT + layerOff + c] = 2.0f * di * di * h[idx] + b[c];
}

__global__ void k_edge(const int* __restrict__ src, const int* __restrict__ dst,
                       const float* __restrict__ dinv, const float* __restrict__ h,
                       float* __restrict__ hcat, int layerOff) {
  int idx = blockIdx.x * blockDim.x + threadIdx.x;   // N_EDGES*8 threads, float4 each
  if (idx >= N_EDGES * 8) return;
  int e = idx >> 3, q = idx & 7;
  int s = src[e], d = dst[e];
  float norm = dinv[s] * dinv[d];
  float4 hv = *(const float4*)(h + (size_t)s * DIM + q * 4);
  float* o = hcat + (size_t)d * CAT + layerOff + q * 4;
  atomicAdd(o + 0, norm * hv.x);
  atomicAdd(o + 1, norm * hv.y);
  atomicAdd(o + 2, norm * hv.z);
  atomicAdd(o + 3, norm * hv.w);
}

__global__ void k_relu_slice(float* __restrict__ hcat, int layerOff) {
  int idx = blockIdx.x * blockDim.x + threadIdx.x;
  if (idx >= N_NODES * DIM) return;
  int r = idx >> 5, c = idx & 31;
  float* p = hcat + (size_t)r * CAT + layerOff + c;
  *p = fmaxf(*p, 0.0f);
}

// ---------------------------------------------------------------- pool + head
__global__ void k_pool(const float* __restrict__ hfc, const int* __restrict__ batch,
                       float* __restrict__ pooled) {
  int idx = blockIdx.x * blockDim.x + threadIdx.x;   // N_NODES * (GDIM/4)
  if (idx >= N_NODES * (GDIM / 4)) return;
  int r = idx >> 5;
  int c = (idx & 31) * 4;
  int g = batch[r];
  float4 v = *(const float4*)(hfc + (size_t)r * GDIM + c);
  float* o = pooled + (size_t)g * GDIM + c;
  atomicAdd(o + 0, v.x);
  atomicAdd(o + 1, v.y);
  atomicAdd(o + 2, v.z);
  atomicAdd(o + 3, v.w);
}

__global__ void k_head(const float* __restrict__ pooled, const float* __restrict__ w,
                       const float* __restrict__ b, float* __restrict__ out) {
  int g = blockIdx.x * blockDim.x + threadIdx.x;
  if (g >= N_GRAPHS) return;
  float l0 = b[0], l1 = b[1];
#pragma unroll 8
  for (int k = 0; k < GDIM; ++k) {
    float p = pooled[(size_t)g * GDIM + k];
    l0 = fmaf(p, w[2 * k + 0], l0);
    l1 = fmaf(p, w[2 * k + 1], l1);
  }
  float m = fmaxf(l0, l1);
  float z = logf(expf(l0 - m) + expf(l1 - m));
  out[2 * g + 0] = l0 - m - z;
  out[2 * g + 1] = l1 - m - z;
}

// ------------------------------------------------------------------------ host
extern "C" void kernel_launch(void* const* d_in, const int* in_sizes, int n_in,
                              void* d_out, int out_size, void* d_ws, size_t ws_size,
                              hipStream_t stream) {
  const float* x    = (const float*)d_in[0];
  const int*   ei   = (const int*)d_in[1];
  const int*   bat  = (const int*)d_in[2];
  const float* w1   = (const float*)d_in[3];
  const float* b1   = (const float*)d_in[4];
  const float* w2   = (const float*)d_in[5];
  const float* b2   = (const float*)d_in[6];
  const float* w3   = (const float*)d_in[7];
  const float* b3   = (const float*)d_in[8];
  const float* fc1w = (const float*)d_in[9];
  const float* fc1b = (const float*)d_in[10];
  const float* fc3w = (const float*)d_in[11];
  const float* fc3b = (const float*)d_in[12];
  float* out = (float*)d_out;
  const int* src = ei;
  const int* dst = ei + N_EDGES;

  char* ws = (char*)d_ws;
  size_t off = 0;
  auto alloc = [&](size_t bytes) -> char* {
    char* p = ws + off;
    off += (bytes + 255) & ~(size_t)255;
    return p;
  };
  float*  deg    = (float*)alloc((size_t)N_NODES * 4);
  float*  dinv   = (float*)alloc((size_t)N_NODES * 4);
  float*  h_tmp  = (float*)alloc((size_t)N_NODES * DIM * 4);
  float*  h_cat  = (float*)alloc((size_t)N_NODES * CAT * 4);
  float*  hfc    = (float*)alloc((size_t)N_NODES * GDIM * 4);
  float*  pooled = (float*)alloc((size_t)N_GRAPHS * GDIM * 4);
  __bf16* xp     = (__bf16*)alloc((size_t)N_NODES * K1PAD * 2);
  __bf16* w1p    = (__bf16*)alloc((size_t)2 * 2 * 512 * 2);
  __bf16* w2p    = (__bf16*)alloc((size_t)1 * 2 * 512 * 2);
  __bf16* w3p    = (__bf16*)alloc((size_t)1 * 2 * 512 * 2);
  __bf16* fc1p   = (__bf16*)alloc((size_t)3 * 8 * 512 * 2);

  hipMemsetAsync(deg, 0, (size_t)N_NODES * 4, stream);
  hipMemsetAsync(pooled, 0, (size_t)N_GRAPHS * GDIM * 4, stream);

  k_deg<<<(N_EDGES + 255) / 256, 256, 0, stream>>>(dst, deg);
  k_dinv<<<(N_NODES + 255) / 256, 256, 0, stream>>>(deg, dinv);

  k_pack_x<<<(N_NODES * (K1PAD / 2) + 255) / 256, 256, 0, stream>>>(x, xp);
  k_pack_b<<<(2048 + 255) / 256, 256, 0, stream>>>(w1, w1p, IN_F, DIM, 2, 2048);
  k_pack_b<<<(1024 + 255) / 256, 256, 0, stream>>>(w2, w2p, DIM, DIM, 1, 1024);
  k_pack_b<<<(1024 + 255) / 256, 256, 0, stream>>>(w3, w3p, DIM, DIM, 1, 1024);
  k_pack_b<<<(12288 + 255) / 256, 256, 0, stream>>>(fc1w, fc1p, CAT, GDIM, 3, 12288);

  const int TILES = (N_NODES + 15) / 16;     // 6250 full M-tiles
  const int GBLKS = (TILES + 7) / 8;         // 8 waves / block
  const int NE_T  = (N_EDGES * 8 + 255) / 256;
  const int ND_T  = (N_NODES * DIM + 255) / 256;

  // layer 1: h = x @ w1  (bf16-packed A, K padded 37->64, branch-free)
  k_gemm_wmma_bf16A<2, 2>
      <<<GBLKS, 256, 0, stream>>>(xp, w1p, h_tmp, DIM, N_NODES);
  k_self_init<<<ND_T, 256, 0, stream>>>(h_tmp, dinv, b1, h_cat, 0);
  k_edge<<<NE_T, 256, 0, stream>>>(src, dst, dinv, h_tmp, h_cat, 0);
  k_relu_slice<<<ND_T, 256, 0, stream>>>(h_cat, 0);

  // layer 2: h = x1 @ w2  (x1 = h_cat cols 0..31, LDA = 96, float4 A-path)
  k_gemm_wmma<2, CAT, 1, false, false>
      <<<GBLKS, 256, 0, stream>>>(h_cat, w2p, nullptr, h_tmp, DIM, N_NODES);
  k_self_init<<<ND_T, 256, 0, stream>>>(h_tmp, dinv, b2, h_cat, DIM);
  k_edge<<<NE_T, 256, 0, stream>>>(src, dst, dinv, h_tmp, h_cat, DIM);

  // layer 3: h = x2 @ w3  (x2 = h_cat cols 32..63)
  k_gemm_wmma<2, CAT, 1, false, false>
      <<<GBLKS, 256, 0, stream>>>(h_cat + DIM, w3p, nullptr, h_tmp, DIM, N_NODES);
  k_self_init<<<ND_T, 256, 0, stream>>>(h_tmp, dinv, b3, h_cat, 2 * DIM);
  k_edge<<<NE_T, 256, 0, stream>>>(src, dst, dinv, h_tmp, h_cat, 2 * DIM);

  // fc1: hfc = relu(h_cat @ fc1w + fc1b)   (concat free; fully unrolled 3-Kstep loop)
  k_gemm_wmma<8, CAT, 3, true, true>
      <<<GBLKS, 256, 0, stream>>>(h_cat, fc1p, fc1b, hfc, GDIM, N_NODES);

  // pool + head
  k_pool<<<(N_NODES * (GDIM / 4) + 255) / 256, 256, 0, stream>>>(hfc, bat, pooled);
  k_head<<<(N_GRAPHS + 255) / 256, 256, 0, stream>>>(pooled, fc3w, fc3b, out);
}
